// FlowerCrossAttention_62191126446225
// MI455X (gfx1250) — matrix operand order repaired and checked by reading, not compiled, not run on hardware
//
#include <hip/hip_runtime.h>

// FlowerCrossAttention for MI455X (gfx1250, wave32, WMMA + TDM).
// Compute-bound on the 4 DxD GEMMs (~494 GFLOP) -> v_wmma_f32_16x16x32_bf16
// (f32 accumulate).  B tiles are streamed into LDS by the Tensor Data Mover
// (tensor_load_to_lds, TENSORcnt) double-buffered against the WMMAs, with
// TDM's pad feature producing the bank-conflict-free LDS row stride.

#define NB 64
#define NSQ 64
#define NSK 384
#define ND 2048
#define NH 16
#define NHD 128

typedef __attribute__((ext_vector_type(16))) __bf16 v16bf;
typedef __attribute__((ext_vector_type(8)))  __bf16 v8bf;
typedef __attribute__((ext_vector_type(8)))  float  v8f;
typedef __attribute__((ext_vector_type(4)))  unsigned int u32x4;
typedef __attribute__((ext_vector_type(8)))  int i32x8;
typedef __attribute__((ext_vector_type(4)))  int i32x4;

__device__ __forceinline__ v16bf mk16(v8bf lo, v8bf hi) {
    v16bf r;
#pragma unroll
    for (int i = 0; i < 8; ++i) { r[i] = lo[i]; r[i + 8] = hi[i]; }
    return r;
}

__device__ __forceinline__ v8f zero8() {
    v8f z;
#pragma unroll
    for (int i = 0; i < 8; ++i) z[i] = 0.0f;
    return z;
}

#define WMMA_BF16(A, B, C) \
    __builtin_amdgcn_wmma_f32_16x16x32_bf16(false, (A), false, (B), (short)0, (C), false, false)

// ---------------------------------------------------------------------------
// TDM: DMA one 32(rows) x 128(cols) bf16 tile, row stride 2048 elements, from
// global into LDS.  pad_enable inserts 4 dwords per 64 data dwords so the LDS
// tile lands with a 136-bf16 row stride (conflict-free B-fragment reads).
// D# built per CDNA5 ISA ch.8 (group0 128b, group1 256b; groups 2/3 unused).
// This toolchain exposes the 6-arg builtin (extra i32x8 group, then cpol).
// ---------------------------------------------------------------------------
__device__ __forceinline__ void tdm_load_b_tile(const __bf16* gsrc, void* lds_dst) {
    const unsigned long long ga = (unsigned long long)(uintptr_t)gsrc;
    const unsigned lds = (unsigned)(uintptr_t)lds_dst;  // low 32 bits = LDS offset
    u32x4 g0;
    g0[0] = 1u;                                   // count=1, user descriptor
    g0[1] = lds;                                  // lds_addr
    g0[2] = (unsigned)(ga & 0xffffffffu);         // global_addr[31:0]
    g0[3] = (unsigned)((ga >> 32) & 0x1ffffffu)   // global_addr[56:32]
          | (2u << 30);                           // type = 2 ("image")
    i32x8 g1;
    g1[0] = (int)((1u << 16)      // data_size = 2 bytes
                | (1u << 20)      // pad_enable
                | (5u << 22)      // pad_interval: 64 dwords (one 128-bf16 row)
                | (3u << 25));    // pad_amount: 4 dwords (8 bf16) -> stride 136
    const unsigned td0 = 2048, td1 = 2048;        // tensor dims (no OOB clipping)
    g1[1] = (int)((td0 & 0xffffu) << 16);                  // tensor_dim0[15:0]
    g1[2] = (int)((td0 >> 16) | ((td1 & 0xffffu) << 16));  // dim0[31:16] | dim1[15:0]
    g1[3] = (int)((td1 >> 16) | (128u << 16));             // dim1[31:16] | tile_dim0=128
    g1[4] = (int)32u;                                      // tile_dim1=32, tile_dim2=0
    const unsigned long long st0 = 2048;                   // tensor_dim0_stride (elems)
    g1[5] = (int)(st0 & 0xffffffffu);
    g1[6] = (int)((st0 >> 32) & 0xffffu);                  // stride hi | dim1_stride lo = 0
    g1[7] = 0;
    i32x4 gz;
    gz[0] = 0; gz[1] = 0; gz[2] = 0; gz[3] = 0;
    i32x8 gz8;
#pragma unroll
    for (int i = 0; i < 8; ++i) gz8[i] = 0;
    __builtin_amdgcn_tensor_load_to_lds(g0, g1, gz, gz, gz8, 0);
}

// ---------------------------------------------------------------------------
// Weight convert + transpose: Wt[k][n] = (bf16) W[n][k].  LDS-tiled 32x32.
// ---------------------------------------------------------------------------
__global__ __launch_bounds__(256) void wtrans(const float* __restrict__ W,
                                              __bf16* __restrict__ Wt) {
    __shared__ float tile[32][33];
    const int kb = blockIdx.x * 32;
    const int nb = blockIdx.y * 32;
    const int tx = threadIdx.x & 31, ty = threadIdx.x >> 5;  // 32 x 8
    for (int i = ty; i < 32; i += 8)
        tile[i][tx] = W[(size_t)(nb + i) * ND + kb + tx];
    __syncthreads();
    for (int i = ty; i < 32; i += 8)
        Wt[(size_t)(kb + i) * ND + nb + tx] = (__bf16)tile[tx][i];
}

// ---------------------------------------------------------------------------
// Tiled WMMA GEMM: C[M x 2048] = A[M x 2048] * Wt[2048 x 2048]  (C = A @ W^T)
// Block = 256 threads = 8 waves; block tile 64(M) x 128(N); K-step 32.
// B tiles arrive by TDM (double-buffered, pad-strided); A tiles are staged by
// VALU (f32->bf16 convert on the fly), also double-buffered.
// One __syncthreads per K-step; wave 0 owns TENSORcnt.
// OUT_S == 0 : write C row-major (bf16 or f32 per OUT_F32)
// OUT_S  > 0 : write bf16 into [b][h][s][hd] with s-extent OUT_S
// ---------------------------------------------------------------------------
template <bool A_F32, bool OUT_F32, int OUT_S>
__global__ __launch_bounds__(256) void gemm_wmma(const void* __restrict__ Aptr,
                                                 const __bf16* __restrict__ Bt,
                                                 void* __restrict__ Cptr, int M) {
    (void)M;
    __shared__ __align__(16) __bf16 As[2][64][48];    // 32 cols + pad
    __shared__ __align__(16) __bf16 Bs[2][32][136];   // 128 cols + TDM pad
    const int t    = threadIdx.x;
    const int lane = t & 31;
    const int w    = t >> 5;
    const int mb   = blockIdx.x * 64;
    const int nbb  = blockIdx.y * 128;
    const int wrow = w & 3;
    const int wc0  = (w >> 2) * 4;

    v8f acc[4];
#pragma unroll
    for (int j = 0; j < 4; ++j) acc[j] = zero8();

    const int ar = t >> 2, ac = (t & 3) * 8;   // A stage: 64 rows x 32 cols

    // ---- prologue: stage tiles for kk=0 ----
    if constexpr (A_F32) {
        const float* src = (const float*)Aptr + (size_t)(mb + ar) * ND + ac;
#pragma unroll
        for (int i = 0; i < 8; ++i) As[0][ar][ac + i] = (__bf16)src[i];
    } else {
        const __bf16* src = (const __bf16*)Aptr + (size_t)(mb + ar) * ND + ac;
#pragma unroll
        for (int i = 0; i < 8; ++i) As[0][ar][ac + i] = src[i];
    }
    if (w == 0) {
        tdm_load_b_tile(Bt + nbb, &Bs[0][0][0]);
        __builtin_amdgcn_s_wait_tensorcnt(0);
    }
    __syncthreads();

    for (int kk = 0; kk < ND; kk += 32) {
        const int cur = (kk >> 5) & 1, nxt = cur ^ 1;
        const bool more = (kk + 32) < ND;

        // ---- kick next tiles while current computes ----
        if (more) {
            if constexpr (A_F32) {
                const float* src = (const float*)Aptr + (size_t)(mb + ar) * ND + kk + 32 + ac;
#pragma unroll
                for (int i = 0; i < 8; ++i) As[nxt][ar][ac + i] = (__bf16)src[i];
            } else {
                const __bf16* src = (const __bf16*)Aptr + (size_t)(mb + ar) * ND + kk + 32 + ac;
#pragma unroll
                for (int i = 0; i < 8; ++i) As[nxt][ar][ac + i] = src[i];
            }
            if (w == 0)
                tdm_load_b_tile(Bt + (size_t)(kk + 32) * ND + nbb, &Bs[nxt][0][0]);
        }

        // ---- A fragment (ISA 16-bit A 16x32 layout) ----
        const int am = wrow * 16 + (lane & 15);
        const int ak = (lane >> 4) * 8;
        v16bf afrag = mk16(*(const v8bf*)&As[cur][am][ak],
                           *(const v8bf*)&As[cur][am][16 + ak]);
#pragma unroll
        for (int j = 0; j < 4; ++j) {
            const int bn = (wc0 + j) * 16;
            v16bf bfrag;
#pragma unroll
            for (int i = 0; i < 8; ++i) {
                bfrag[i] = Bs[cur][ak + i][bn + (lane & 15)];
                bfrag[i + 8] = Bs[cur][16 + ak + i][bn + (lane & 15)];
            }
            acc[j] = WMMA_BF16(afrag, bfrag, acc[j]);
        }

        if (more && w == 0) __builtin_amdgcn_s_wait_tensorcnt(0);
        __syncthreads();
    }

    // ---- epilogue ----
#pragma unroll
    for (int j = 0; j < 4; ++j) {
        const int n = nbb + (wc0 + j) * 16 + (lane & 15);
#pragma unroll
        for (int i = 0; i < 8; ++i) {
            const int m = mb + wrow * 16 + i + 8 * (lane >> 4);
            const float v = acc[j][i];
            if constexpr (OUT_F32) {
                ((float*)Cptr)[(size_t)m * ND + n] = v;
            } else if constexpr (OUT_S == 0) {
                ((__bf16*)Cptr)[(size_t)m * ND + n] = (__bf16)v;
            } else {
                const int b = m / OUT_S, s = m % OUT_S;
                const int h = n >> 7, d = n & 127;
                ((__bf16*)Cptr)[((((size_t)b * NH + h) * OUT_S + s) << 7) + d] = (__bf16)v;
            }
        }
    }
}

// ---------------------------------------------------------------------------
// RMSNorm + RoPE over one head-row of 128.  One 128-thread block per row.
// MODE 0: Q, rows [bh][s<64][128], write in-place same layout.
// MODE 1: K, rows [bh][s<384][128], write transposed [bh][hd][384].
// ---------------------------------------------------------------------------
template <int MODE>
__global__ __launch_bounds__(128) void normrope(const __bf16* in, __bf16* out,
                                                const float* __restrict__ nw,
                                                const float* __restrict__ cosb,
                                                const float* __restrict__ sinb) {
    constexpr int S = (MODE == 0) ? NSQ : NSK;
    const int bid = blockIdx.x;  // bh * S + s
    const int bh = bid / S, s = bid % S;
    const int d = threadIdx.x;
    __shared__ float sh[128];
    __shared__ float red[4];

    float v = (float)in[((size_t)bid << 7) + d];
    float sq = v * v;
#pragma unroll
    for (int m = 1; m < 32; m <<= 1) sq += __shfl_xor(sq, m, 32);
    if ((d & 31) == 0) red[d >> 5] = sq;
    __syncthreads();
    const float tot = red[0] + red[1] + red[2] + red[3];
    const float rms = rsqrtf(tot * (1.0f / 128.0f) + 1e-6f);
    const float nv = v * rms * nw[d];
    sh[d] = nv;
    __syncthreads();
    const int dd = d & 63;
    const float c = cosb[s * 64 + dd];
    const float sn = sinb[s * 64 + dd];
    const float other = sh[d ^ 64];
    const float o = (d < 64) ? (nv * c - other * sn) : (nv * c + other * sn);
    if constexpr (MODE == 0) {
        out[((size_t)bid << 7) + d] = (__bf16)o;
    } else {
        out[(((size_t)bh << 7) + d) * NSK + s] = (__bf16)o;
    }
}

// ---------------------------------------------------------------------------
// Flash-style attention per (b,h).  128 threads = 4 waves; wave w owns query
// rows 16w..16w+15.  Online softmax; scores and P*V on WMMA.
// ---------------------------------------------------------------------------
__global__ __launch_bounds__(128) void attn(const __bf16* __restrict__ Qp,   // [bh][64][128]
                                            const __bf16* __restrict__ Ktp,  // [bh][128][384]
                                            const __bf16* __restrict__ Vp,   // [bh][384][128]
                                            const unsigned char* __restrict__ mask, // [B][384]
                                            __bf16* __restrict__ Y) {
    const int bh = blockIdx.x;
    const int b = bh >> 4, h = bh & 15;
    const int t = threadIdx.x, lane = t & 31, w = t >> 5;
    const float scale = 0.08838834764831845f;  // 128^-0.5
    __shared__ __align__(16) __bf16 Ps[4][16][48];  // per-wave P bounce

    const __bf16* Qb = Qp + ((size_t)bh << 13);
    const __bf16* Kb = Ktp + (size_t)bh * (NHD * NSK);
    const __bf16* Vb = Vp + (size_t)bh * (NSK * NHD);
    const unsigned char* mk = mask + b * NSK;

    v16bf qf[4];
    {
        const __bf16* qrow = Qb + (size_t)(w * 16 + (lane & 15)) * NHD;
#pragma unroll
        for (int c = 0; c < 4; ++c) {
            const int k0 = c * 32 + ((lane >> 4) << 3);
            qf[c] = mk16(*(const v8bf*)(qrow + k0), *(const v8bf*)(qrow + k0 + 16));
        }
    }

    v8f acc[8];
#pragma unroll
    for (int tt = 0; tt < 8; ++tt) acc[tt] = zero8();
    float rmax[8], rsum[8];
#pragma unroll
    for (int i = 0; i < 8; ++i) { rmax[i] = -1e30f; rsum[i] = 0.0f; }

    for (int nb = 0; nb < NSK; nb += 32) {
        v8f s0 = zero8(), s1 = zero8();
#pragma unroll
        for (int c = 0; c < 4; ++c) {
            const __bf16* kr = Kb + (size_t)(c * 32 + ((lane >> 4) << 3)) * NSK + nb + (lane & 15);
            v16bf kf0, kf1;
#pragma unroll
            for (int i = 0; i < 8; ++i) {
                kf0[i] = kr[i * NSK];
                kf0[i + 8] = kr[(16 + i) * NSK];
                kf1[i] = kr[i * NSK + 16];
                kf1[i + 8] = kr[(16 + i) * NSK + 16];
            }
            s0 = WMMA_BF16(qf[c], kf0, s0);
            s1 = WMMA_BF16(qf[c], kf1, s1);
        }
        const bool m0 = mk[nb + (lane & 15)] != 0;
        const bool m1 = mk[nb + 16 + (lane & 15)] != 0;
#pragma unroll
        for (int i = 0; i < 8; ++i) {
            s0[i] = m0 ? s0[i] * scale : -1e30f;
            s1[i] = m1 ? s1[i] * scale : -1e30f;
        }
        float corr[8];
#pragma unroll
        for (int i = 0; i < 8; ++i) {
            float tm = fmaxf(s0[i], s1[i]);
#pragma unroll
            for (int m = 1; m < 16; m <<= 1) tm = fmaxf(tm, __shfl_xor(tm, m, 32));
            const float nm = fmaxf(rmax[i], tm);
            const float cr = __expf(rmax[i] - nm);
            const float p0 = __expf(s0[i] - nm);
            const float p1 = __expf(s1[i] - nm);
            float ts = p0 + p1;
#pragma unroll
            for (int m = 1; m < 16; m <<= 1) ts += __shfl_xor(ts, m, 32);
            rsum[i] = rsum[i] * cr + ts;
            rmax[i] = nm;
            corr[i] = cr;
            s0[i] = p0;
            s1[i] = p1;
        }
#pragma unroll
        for (int tt = 0; tt < 8; ++tt) {
#pragma unroll
            for (int i = 0; i < 8; ++i) acc[tt][i] *= corr[i];
        }
#pragma unroll
        for (int i = 0; i < 8; ++i) {
            const int m = i + ((lane >> 4) << 3);
            Ps[w][m][lane & 15] = (__bf16)s0[i];
            Ps[w][m][16 + (lane & 15)] = (__bf16)s1[i];
        }
        v16bf pf;
        {
            const int k0 = (lane >> 4) << 3;
            pf = mk16(*(const v8bf*)&Ps[w][lane & 15][k0],
                      *(const v8bf*)&Ps[w][lane & 15][16 + k0]);
        }
#pragma unroll
        for (int tt = 0; tt < 8; ++tt) {
            const __bf16* vr = Vb + (size_t)(nb + ((lane >> 4) << 3)) * NHD + tt * 16 + (lane & 15);
            v16bf vf;
#pragma unroll
            for (int i = 0; i < 8; ++i) {
                vf[i] = vr[i * NHD];
                vf[i + 8] = vr[(16 + i) * NHD];
            }
            acc[tt] = WMMA_BF16(pf, vf, acc[tt]);
        }
    }

#pragma unroll
    for (int i = 0; i < 8; ++i) rsum[i] = 1.0f / rsum[i];
#pragma unroll
    for (int tt = 0; tt < 8; ++tt) {
        const int dcol = h * NHD + tt * 16 + (lane & 15);
#pragma unroll
        for (int i = 0; i < 8; ++i) {
            const int m = w * 16 + i + ((lane >> 4) << 3);
            Y[((size_t)b * NSQ + m) * ND + dcol] = (__bf16)(acc[tt][i] * rsum[i]);
        }
    }
}

// ---------------------------------------------------------------------------
// Workspace layout (bytes), total ~369 MB.
// ---------------------------------------------------------------------------
extern "C" void kernel_launch(void* const* d_in, const int* in_sizes, int n_in,
                              void* d_out, int out_size, void* d_ws, size_t ws_size,
                              hipStream_t stream) {
    (void)in_sizes; (void)n_in; (void)out_size; (void)ws_size;
    const float* x    = (const float*)d_in[0];
    const float* ctx  = (const float*)d_in[1];
    const unsigned char* mask = (const unsigned char*)d_in[2];  // jax bool -> 1 byte
    const float* wq   = (const float*)d_in[3];
    const float* wk   = (const float*)d_in[4];
    const float* wv   = (const float*)d_in[5];
    const float* wo   = (const float*)d_in[6];
    const float* qnw  = (const float*)d_in[7];
    const float* knw  = (const float*)d_in[8];
    const float* qcos = (const float*)d_in[9];
    const float* qsin = (const float*)d_in[10];
    const float* kcos = (const float*)d_in[11];
    const float* ksin = (const float*)d_in[12];

    char* ws = (char*)d_ws;
    const size_t WB = (size_t)ND * ND * 2;  // 8,388,608
    __bf16* wq_t = (__bf16*)(ws);
    __bf16* wk_t = (__bf16*)(ws + WB);
    __bf16* wv_t = (__bf16*)(ws + 2 * WB);
    __bf16* wo_t = (__bf16*)(ws + 3 * WB);
    __bf16* Qb   = (__bf16*)(ws + 4 * WB);                                 // 16.8 MB
    __bf16* Kb   = (__bf16*)(ws + 4 * WB + 16777216);                      // 100.7 MB
    __bf16* Vb   = (__bf16*)(ws + 4 * WB + 16777216 + 100663296);          // 100.7 MB
    __bf16* Ktp  = (__bf16*)(ws + 4 * WB + 16777216 + 2ull * 100663296);   // 100.7 MB
    __bf16* Ybf  = (__bf16*)(ws + 4 * WB + 16777216 + 3ull * 100663296);   // 16.8 MB

    // 1) weights -> bf16, transposed
    wtrans<<<dim3(64, 64), 256, 0, stream>>>(wq, wq_t);
    wtrans<<<dim3(64, 64), 256, 0, stream>>>(wk, wk_t);
    wtrans<<<dim3(64, 64), 256, 0, stream>>>(wv, wv_t);
    wtrans<<<dim3(64, 64), 256, 0, stream>>>(wo, wo_t);

    // 2) projections (WMMA + TDM), K/V written directly in [b][h][s][hd]
    gemm_wmma<true, false, NSQ><<<dim3(64, 16), 256, 0, stream>>>(x, wq_t, Qb, NB * NSQ);
    gemm_wmma<true, false, NSK><<<dim3(384, 16), 256, 0, stream>>>(ctx, wk_t, Kb, NB * NSK);
    gemm_wmma<true, false, NSK><<<dim3(384, 16), 256, 0, stream>>>(ctx, wv_t, Vb, NB * NSK);

    // 3) RMSNorm + RoPE (Q in place; K -> transposed layout)
    normrope<0><<<NB * NH * NSQ, 128, 0, stream>>>(Qb, Qb, qnw, qcos, qsin);
    normrope<1><<<NB * NH * NSK, 128, 0, stream>>>(Kb, Ktp, knw, kcos, ksin);

    // 4) attention (WMMA flash)
    attn<<<NB * NH, 128, 0, stream>>>(Qb, Ktp, Vb, mask, Ybf);

    // 5) output projection -> f32 d_out
    gemm_wmma<false, true, 0><<<dim3(64, 16), 256, 0, stream>>>(Ybf, wo_t, d_out, NB * NSQ);
}
